// GAT_87153476370807
// MI455X (gfx1250) — matrix operand order repaired and checked by reading, not compiled, not run on hardware
//
#include <hip/hip_runtime.h>
#include <hip/hip_bf16.h>
#include <limits.h>

// ---------------------------------------------------------------------------
// GATv2 x4 layers, N=50000, E=800000, F=128, HID=128, HEADS=2, OUT=1
// Dense transforms: async-to-LDS staged operands + v_wmma_f32_16x16x32_f16.
// Sparse phase: L2-resident gathers + float atomics (wave32 layout).
// ---------------------------------------------------------------------------

typedef __attribute__((ext_vector_type(16))) _Float16 v16h;
typedef __attribute__((ext_vector_type(8)))  _Float16 v8h;
typedef __attribute__((ext_vector_type(8)))  float    v8f;
typedef __attribute__((vector_size(16)))     int      v4i_vs;   // b128 payload

#define GN 50000
#define GE 800000
#define GF 128      // feature width / HID
#define GH 2        // heads
#define HC 256      // HEADS*HID

// ---- CDNA5 async global->LDS copy (ASYNCcnt path), with safe fallback -----
#if __has_builtin(__builtin_amdgcn_global_load_async_to_lds_b128)
#define ASYNC_CP16(gp, lp)                                                     \
    __builtin_amdgcn_global_load_async_to_lds_b128(                            \
        (v4i_vs*)(void*)(gp),                                                  \
        (__attribute__((address_space(3))) v4i_vs*)(void*)(lp), 0, 0)
#else
#define ASYNC_CP16(gp, lp) (*(v8h*)(lp) = *(const v8h*)(gp))
#endif

#if __has_builtin(__builtin_amdgcn_s_wait_asynccnt)
#define ASYNC_WAIT() __builtin_amdgcn_s_wait_asynccnt(0)
#elif __has_builtin(__builtin_amdgcn_global_load_async_to_lds_b128)
#define ASYNC_WAIT() asm volatile("s_wait_asynccnt 0x0" ::: "memory")
#else
#define ASYNC_WAIT() ((void)0)
#endif

// ---------------- initial conversion x -> f16, root idx init ---------------
__global__ void k_convert_x(const float* __restrict__ x,
                            _Float16* __restrict__ h16,
                            int* __restrict__ rootidx) {
    int i = blockIdx.x * blockDim.x + threadIdx.x;
    if (i == 0) *rootidx = INT_MAX;
    if (i < GN * GF) h16[i] = (_Float16)x[i];
}

__global__ void k_root_scan(const float* __restrict__ x, int* __restrict__ rootidx) {
    int i = blockIdx.x * blockDim.x + threadIdx.x;
    if (i < GN && x[(size_t)i * GF] == 0.0f) atomicMin(rootidx, i);
}

// -------- weight conversion: W[l] f32 [128,256] -> Wt16 f16 [256,128] ------
__global__ void k_conv_w(const float* __restrict__ Wl,
                         const float* __restrict__ Wr,
                         _Float16* __restrict__ Wt16, int layer) {
    int i = blockIdx.x * blockDim.x + threadIdx.x;     // 2 * 256 * 128
    if (i >= 2 * HC * GF) return;
    int which = i / (HC * GF);
    int r = i - which * (HC * GF);
    int n = r / GF, k = r - n * GF;
    const float* W = (which ? Wr : Wl) + (size_t)layer * GF * HC;
    Wt16[(size_t)which * HC * GF + (size_t)n * GF + k] = (_Float16)W[(size_t)k * HC + n];
}

// ---------------- WMMA GEMM: [N,128]f16 @ [128,256]f16 + bias -> f32 -------
// grid = (3125 mt, 2 which, 2 nhalf), block = 256 (8 waves).
// Per block: async-stage 128 weight columns (32KB) + 16-row A strip (4KB)
// into LDS; each wave computes one 16x16 tile from LDS operands.
__global__ void __launch_bounds__(256)
k_gemm(const _Float16* __restrict__ h16,
       const _Float16* __restrict__ Wt16,   // [2][256][128] (col-major W)
       const float* __restrict__ bl,
       const float* __restrict__ br,
       float* __restrict__ XL, float* __restrict__ XR) {
    __shared__ __attribute__((aligned(32))) _Float16 sW[128 * GF]; // 32 KB
    __shared__ __attribute__((aligned(32))) _Float16 sA[16 * GF];  //  4 KB

    const int tid   = threadIdx.x;
    const int mt    = blockIdx.x;
    const int which = blockIdx.y;
    const int zh    = blockIdx.z;            // column half: 0 -> cols 0..127

    const _Float16* gW = Wt16 + (size_t)which * HC * GF + (size_t)zh * 128 * GF;
    const _Float16* gA = h16 + (size_t)mt * 16 * GF;

    // stage B: 16384 halfs = 2048 x b128, 8 per thread
#pragma unroll
    for (int j = 0; j < 8; ++j) {
        int off = (tid + j * 256) * 8;       // halfs
        ASYNC_CP16(gW + off, sW + off);
    }
    // stage A strip: 2048 halfs = 256 x b128, 1 per thread
    {
        int off = tid * 8;
        ASYNC_CP16(gA + off, sA + off);
    }
    ASYNC_WAIT();
    __syncthreads();

    const int wave  = tid >> 5;
    const int lane  = tid & 31;
    const int l16   = lane & 15;
    const int khalf = lane >> 4;             // 0 or 1

    const _Float16* arow = sA + l16 * GF;                 // A: M = lane%16
    const _Float16* bcol = sW + (wave * 16 + l16) * GF;   // B: N = lane%16

    v8f acc = {};
#pragma unroll
    for (int k0 = 0; k0 < GF; k0 += 32) {
        // A fragment: lanes 0-15 K = k0..k0+15, lanes 16-31 +16
        v16h a = *(const v16h*)(arow + k0 + khalf * 16);
        // B fragment: lanes 0-15 K = k0..k0+15, lanes 16-31 +16
        v16h b = *(const v16h*)(bcol + k0 + khalf * 16);
        acc = __builtin_amdgcn_wmma_f32_16x16x32_f16(
            false, a, false, b, (short)0, acc, false, false);
    }

    // D layout: VGPR v -> M = khalf*8 + v, N = lane%16
    const int col  = zh * 128 + wave * 16 + l16;
    const int orow = mt * 16 + khalf * 8;
    const float* bias = which ? br : bl;
    float* C = which ? XR : XL;
    const float bv = bias[col];
#pragma unroll
    for (int v = 0; v < 8; ++v)
        C[(size_t)(orow + v) * HC + col] = acc[v] + bv;
}

// ---------------- per-layer accumulator init -------------------------------
__global__ void k_init(float* __restrict__ acc, unsigned* __restrict__ mkey,
                       float* __restrict__ denom) {
    int i = blockIdx.x * blockDim.x + threadIdx.x;
    if (i < GN * HC) acc[i] = 0.0f;
    if (i < GN * GH) { mkey[i] = 0u; denom[i] = 0.0f; }
}

// --------- edge logits: one wave per edge, 16 lanes per head ---------------
__global__ void k_edge_logits(const long long* __restrict__ ei,
                              const float* __restrict__ XL,
                              const float* __restrict__ XR,
                              const float* __restrict__ att,   // [2,128]
                              float* __restrict__ elog,
                              unsigned* __restrict__ mkey) {
    int wid = (blockIdx.x * blockDim.x + threadIdx.x) >> 5;
    if (wid >= GE) return;
    int lane = threadIdx.x & 31;
    int src = (int)ei[wid];
    int dst = (int)ei[GE + wid];
    int head = lane >> 4;
    int c0 = (lane & 15) * 8;
    const float* pl = XL + (size_t)src * HC + head * GF + c0;
    const float* pr = XR + (size_t)dst * HC + head * GF + c0;
    const float* pa = att + head * GF + c0;
    float s = 0.0f;
#pragma unroll
    for (int i = 0; i < 8; ++i) {
        float v = pl[i] + pr[i];
        v = v > 0.0f ? v : 0.2f * v;        // leaky relu
        s = fmaf(pa[i], v, s);
    }
    // reduce across the 16-lane half (wave32)
    s += __shfl_xor(s, 1, 32);
    s += __shfl_xor(s, 2, 32);
    s += __shfl_xor(s, 4, 32);
    s += __shfl_xor(s, 8, 32);
    if ((lane & 15) == 0) {
        elog[(size_t)wid * GH + head] = s;
        unsigned u = __float_as_uint(s);
        unsigned key = (u & 0x80000000u) ? ~u : (u | 0x80000000u); // monotone
        atomicMax(&mkey[(size_t)dst * GH + head], key);
    }
}

// ---------------- decode monotone max keys back to float (in place) --------
__global__ void k_decode_m(unsigned* __restrict__ mkey) {
    int i = blockIdx.x * blockDim.x + threadIdx.x;
    if (i >= GN * GH) return;
    unsigned key = mkey[i];
    float m;
    if (key == 0u)                 m = 0.0f;                       // no edges
    else if (key & 0x80000000u)    m = __uint_as_float(key & 0x7fffffffu);
    else                           m = __uint_as_float(~key);
    ((float*)mkey)[i] = m;
}

// ---------------- p = exp(e - m[dst]); denom[dst] += p ---------------------
__global__ void k_exp_sum(const long long* __restrict__ ei,
                          float* __restrict__ elog,
                          const float* __restrict__ mf,
                          float* __restrict__ denom) {
    int i = blockIdx.x * blockDim.x + threadIdx.x;
    if (i >= GE * GH) return;
    int eidx = i >> 1, h = i & 1;
    int dst = (int)ei[GE + eidx];
    float p = __expf(elog[i] - mf[dst * GH + h]);
    elog[i] = p;
    atomicAdd(&denom[dst * GH + h], p);
}

// ------- scatter: acc[dst] += alpha * xl[src]; 64 threads per edge ---------
__global__ void k_scatter(const long long* __restrict__ ei,
                          const float* __restrict__ ep,
                          const float* __restrict__ denom,
                          const float* __restrict__ XL,
                          float* __restrict__ acc) {
    int i = blockIdx.x * blockDim.x + threadIdx.x;   // GE*64 = 51.2M < 2^31
    if (i >= GE * 64) return;
    int g = i & 63;
    int eidx = i >> 6;
    int h  = g >> 5;
    int c4 = (g & 31) * 4;
    int src = (int)ei[eidx];
    int dst = (int)ei[GE + eidx];
    float p = ep[(size_t)eidx * GH + h];
    float d = denom[dst * GH + h];
    float alpha = p / (d + 1e-16f);
    const float4 xv = *(const float4*)(XL + (size_t)src * HC + h * GF + c4);
    float* o = acc + (size_t)dst * HC + h * GF + c4;
    atomicAdd(o + 0, alpha * xv.x);
    atomicAdd(o + 1, alpha * xv.y);
    atomicAdd(o + 2, alpha * xv.z);
    atomicAdd(o + 3, alpha * xv.w);
}

// -------- finalize: head mean + bias + relu -> hbuf (f32) and h16 ----------
__global__ void k_finalize(const float* __restrict__ acc,
                           const float* __restrict__ bias,  // [128]
                           float* __restrict__ hbuf,
                           _Float16* __restrict__ h16) {
    int i = blockIdx.x * blockDim.x + threadIdx.x;
    if (i >= GN * GF) return;
    int n = i >> 7, c = i & 127;
    float v = 0.5f * (acc[(size_t)n * HC + c] + acc[(size_t)n * HC + GF + c]) + bias[c];
    v = fmaxf(v, 0.0f);
    hbuf[i] = v;
    h16[i] = (_Float16)v;
}

// ---------------- final FC + sigmoid on the root node ----------------------
__global__ void k_fc(const float* __restrict__ hbuf,
                     const float* __restrict__ fcW,  // [128,1]
                     const float* __restrict__ fcb,  // [1]
                     const int* __restrict__ rootidx,
                     float* __restrict__ out) {
    __shared__ float red[128];
    int tid = threadIdx.x;
    int r = *rootidx;
    if (r < 0 || r >= GN) r = 0;
    red[tid] = hbuf[(size_t)r * GF + tid] * fcW[tid];
    __syncthreads();
    for (int s = 64; s > 0; s >>= 1) {
        if (tid < s) red[tid] += red[tid + s];
        __syncthreads();
    }
    if (tid == 0) {
        float z = red[0] + fcb[0];
        out[0] = 1.0f / (1.0f + __expf(-z));
    }
}

// ---------------------------------------------------------------------------
extern "C" void kernel_launch(void* const* d_in, const int* in_sizes, int n_in,
                              void* d_out, int out_size, void* d_ws, size_t ws_size,
                              hipStream_t stream) {
    (void)in_sizes; (void)n_in; (void)out_size; (void)ws_size;
    const float*      x    = (const float*)d_in[0];
    const long long*  ei   = (const long long*)d_in[1];
    const float*      Wl   = (const float*)d_in[2];
    const float*      bl   = (const float*)d_in[3];
    const float*      Wr   = (const float*)d_in[4];
    const float*      br   = (const float*)d_in[5];
    const float*      att  = (const float*)d_in[6];
    const float*      bias = (const float*)d_in[7];
    const float*      fcW  = (const float*)d_in[8];
    const float*      fcb  = (const float*)d_in[9];
    float* out = (float*)d_out;

    // ---- carve workspace ----
    char* p = (char*)d_ws;
    auto carve = [&](size_t bytes) -> char* {
        char* r = p; p += (bytes + 255) & ~(size_t)255; return r;
    };
    _Float16* h16   = (_Float16*)carve((size_t)GN * GF * 2);   // 12.8 MB
    _Float16* Wt16  = (_Float16*)carve((size_t)2 * HC * GF * 2);
    float*    XL    = (float*)carve((size_t)GN * HC * 4);      // 51.2 MB
    float*    XR    = (float*)carve((size_t)GN * HC * 4);      // 51.2 MB
    float*    acc   = (float*)carve((size_t)GN * HC * 4);      // 51.2 MB
    float*    hbuf  = (float*)carve((size_t)GN * GF * 4);      // 25.6 MB
    float*    elog  = (float*)carve((size_t)GE * GH * 4);      // 6.4 MB
    unsigned* mkey  = (unsigned*)carve((size_t)GN * GH * 4);
    float*    denom = (float*)carve((size_t)GN * GH * 4);
    int*      root  = (int*)carve(256);

    const int B = 256;
    k_convert_x<<<(GN * GF + B - 1) / B, B, 0, stream>>>(x, h16, root);
    k_root_scan<<<(GN + B - 1) / B, B, 0, stream>>>(x, root);

    for (int l = 0; l < 4; ++l) {
        k_conv_w<<<(2 * HC * GF + B - 1) / B, B, 0, stream>>>(Wl, Wr, Wt16, l);
        k_gemm<<<dim3(GN / 16, 2, 2), B, 0, stream>>>(
            h16, Wt16, bl + (size_t)l * HC, br + (size_t)l * HC, XL, XR);
        k_init<<<(GN * HC + B - 1) / B, B, 0, stream>>>(acc, mkey, denom);
        k_edge_logits<<<GE / 8, B, 0, stream>>>(ei, XL, XR,
                                                att + (size_t)l * GH * GF, elog, mkey);
        k_decode_m<<<(GN * GH + B - 1) / B, B, 0, stream>>>(mkey);
        k_exp_sum<<<(GE * GH + B - 1) / B, B, 0, stream>>>(ei, elog, (float*)mkey, denom);
        k_scatter<<<GE * 64 / B, B, 0, stream>>>(ei, elog, denom, XL, acc);
        k_finalize<<<(GN * GF + B - 1) / B, B, 0, stream>>>(acc, bias + (size_t)l * GF,
                                                            hbuf, h16);
    }
    k_fc<<<1, 128, 0, stream>>>(hbuf, fcW, fcb, root, out);
}
